// CASSBlock_35622458753210
// MI455X (gfx1250) — matrix operand order repaired and checked by reading, not compiled, not run on hardware
//
#include <hip/hip_runtime.h>
#include <hip/hip_bf16.h>

// ---------------- problem constants ----------------
#define BB   32
#define HH   56
#define WW   56
#define CC   192
#define DD   384          // D_INNER = 2*C
#define LL   (HH*WW)      // 3136
#define MM   (BB*LL)      // 100352 rows
#define MT   (MM/16)      // 6272 row tiles
#define EPSV 1e-5f

#define KP1  (CC + 8)     // padded LDS row stride (halves), fc1 B slab

typedef __attribute__((ext_vector_type(16))) _Float16 v16h;
typedef __attribute__((ext_vector_type(8)))  _Float16 v8h;
typedef __attribute__((ext_vector_type(8)))  float    v8f;
typedef __attribute__((ext_vector_type(4)))  int      v4i;

// ---------------- CDNA5 async global->LDS copy (guarded) ----------------
#if defined(__has_builtin)
#if __has_builtin(__builtin_amdgcn_global_load_async_to_lds_b128) && \
    __has_builtin(__builtin_amdgcn_s_wait_asynccnt)
#define USE_ASYNC_LDS 1
#endif
#if __has_builtin(__builtin_amdgcn_sched_barrier)
#define SCHED_FENCE() __builtin_amdgcn_sched_barrier(0)
#endif
#endif
#ifndef SCHED_FENCE
#define SCHED_FENCE()
#endif

typedef __attribute__((address_space(1))) v4i g_v4i;
typedef __attribute__((address_space(3))) v4i l_v4i;

__device__ __forceinline__ void cp_b128(const _Float16* g, _Float16* l) {
#ifdef USE_ASYNC_LDS
    __builtin_amdgcn_global_load_async_to_lds_b128((g_v4i*)g, (l_v4i*)l, 0, 0);
#else
    *(v8h*)l = *(const v8h*)g;
#endif
}

__device__ __forceinline__ void stage_wait() {
#ifdef USE_ASYNC_LDS
    __builtin_amdgcn_s_wait_asynccnt(0);
#endif
    __syncthreads();
}

// ---------------- workspace layout (bytes) ----------------
#define OFF_XN   ((size_t)0)
#define OFF_GRAY (OFF_XN   + (size_t)MM*CC*2)
#define OFF_GSUM (OFF_GRAY + (size_t)MM*4)
#define OFF_ISV  (OFF_GSUM + (size_t)256)
#define OFF_W1T  (OFF_ISV  + (size_t)256)
#define OFF_W2T  (OFF_W1T  + (size_t)DD*CC*2)
#define OFF_T1   (OFF_W2T  + (size_t)CC*DD*2)

// ---------------- K0: LayerNorm + gray ----------------
__global__ void ln_kernel(const float* __restrict__ x,
                          const float* __restrict__ gamma,
                          const float* __restrict__ beta,
                          _Float16* __restrict__ xn,
                          float* __restrict__ gray) {
    const int lane = threadIdx.x & 31;
    const int wave = threadIdx.x >> 5;
    const int row  = blockIdx.x * 8 + wave;
    if (row >= MM) return;
    const float* px = x + (size_t)row * CC;

    float v[6];
    float s = 0.f;
#pragma unroll
    for (int j = 0; j < 6; ++j) { v[j] = px[lane + 32*j]; s += v[j]; }
#pragma unroll
    for (int o = 16; o > 0; o >>= 1) s += __shfl_xor(s, o, 32);
    const float mu = s * (1.0f/CC);

    float vs = 0.f;
#pragma unroll
    for (int j = 0; j < 6; ++j) { float d = v[j] - mu; vs += d*d; }
#pragma unroll
    for (int o = 16; o > 0; o >>= 1) vs += __shfl_xor(vs, o, 32);
    const float rstd = rsqrtf(vs * (1.0f/CC) + EPSV);

    float gs = 0.f;
    _Float16* po = xn + (size_t)row * CC;
#pragma unroll
    for (int j = 0; j < 6; ++j) {
        const int c = lane + 32*j;
        const float y = (v[j] - mu) * rstd * gamma[c] + beta[c];
        po[c] = (_Float16)y;
        gs += y;
    }
#pragma unroll
    for (int o = 16; o > 0; o >>= 1) gs += __shfl_xor(gs, o, 32);
    if (lane == 0) gray[row] = gs * (1.0f/CC);
}

// ---------------- K1: per-batch gradient sums ----------------
__global__ void grad_kernel(const float* __restrict__ gray,
                            float* __restrict__ gsum) {
    __shared__ float sh[256], sv[256];
    const int b = blockIdx.y;
    const int t = blockIdx.x * 256 + threadIdx.x;
    float gh = 0.f, gv = 0.f;
    if (t < LL) {
        const float* g = gray + (size_t)b * LL;
        const int h = t / WW, w = t - h * WW;
        const float c = g[t];
        if (w < WW-1) gh = fabsf(g[t+1]   - c);
        if (h < HH-1) gv = fabsf(g[t+WW]  - c);
    }
    sh[threadIdx.x] = gh; sv[threadIdx.x] = gv;
    __syncthreads();
    for (int o = 128; o > 0; o >>= 1) {
        if (threadIdx.x < o) {
            sh[threadIdx.x] += sh[threadIdx.x + o];
            sv[threadIdx.x] += sv[threadIdx.x + o];
        }
        __syncthreads();
    }
    if (threadIdx.x == 0) {
        atomicAdd(&gsum[b*2+0], sh[0]);
        atomicAdd(&gsum[b*2+1], sv[0]);
    }
}

// ---------------- K2: selector MLP + argmax ----------------
__global__ void sel_kernel(const float* __restrict__ gsum,
                           const float* __restrict__ w1,
                           const float* __restrict__ b1,
                           const float* __restrict__ w2,
                           const float* __restrict__ b2,
                           int* __restrict__ isv) {
    const int b = threadIdx.x;
    if (b >= BB) return;
    const float gh = gsum[b*2+0] * (1.0f / (HH * (WW-1)));
    const float gv = gsum[b*2+1] * (1.0f / ((HH-1) * WW));
    float s[4];
    s[0] = gh; s[1] = gv; s[2] = 0.4f*(gh+gv); s[3] = fabsf(gh-gv);
    float hl[16];
#pragma unroll
    for (int j = 0; j < 16; ++j) {
        float t = b1[j];
#pragma unroll
        for (int i = 0; i < 4; ++i) t += s[i] * w1[i*16+j];
        hl[j] = t > 0.f ? t : 0.f;
    }
    float best = -1e30f; int idx = 0;
#pragma unroll
    for (int j = 0; j < 4; ++j) {
        float t = b2[j];
#pragma unroll
        for (int i = 0; i < 16; ++i) t += hl[i] * w2[i*4+j];
        if (t > best) { best = t; idx = j; }   // softmax is monotonic
    }
    isv[b] = (idx == 1) ? 1 : 0;
}

// ---------------- K_pre: weight transpose + f32->f16 ----------------
__global__ void wt_kernel(const float* __restrict__ w, _Float16* __restrict__ wt,
                          int K, int N) {
    const int i = blockIdx.x * 256 + threadIdx.x;
    if (i < K * N) {
        const int k = i / N, n = i - k * N;
        wt[(size_t)n * K + k] = (_Float16)w[i];
    }
}

// ---------------- WMMA fragment helpers ----------------
__device__ __forceinline__ v16h make_frag(const v8h lo, const v8h hi) {
    v16h a;
#pragma unroll
    for (int i = 0; i < 8; ++i) { a[i] = lo[i]; a[i+8] = hi[i]; }
    return a;
}

// B fragment: lane holds column n, K values kb..kb+7 and kb+16..kb+23 (two b128 loads).
__device__ __forceinline__ v16h load_b_frag(const _Float16* base, int n,
                                            int stride, int kb) {
    const _Float16* p = base + (size_t)n * stride + kb;
    const v8h lo = *(const v8h*)p;
    const v8h hi = *(const v8h*)(p + 16);
    return make_frag(lo, hi);
}

// ---------------- K3: fc1 GEMM (wave tile 16x128, B staged in LDS) -------
// grid: (MT/8, DD/128=3), 256 threads = 8 waves.
// __launch_bounds__(256, 4): min 4 waves/EU -> ~256 VGPR budget, no spills.
__global__ void __launch_bounds__(256, 4)
fc1_kernel(const _Float16* __restrict__ xn,
           const _Float16* __restrict__ w1t,
           const float* __restrict__ b1,
           const int* __restrict__ isv,
           _Float16* __restrict__ t1) {
    __shared__ _Float16 sB[128 * KP1];          // 51,200 B
    const int lane   = threadIdx.x & 31;
    const int wave   = threadIdx.x >> 5;
    const int m_tile = blockIdx.x * 8 + wave;
    const int n_base = blockIdx.y * 128;

    // stage 128 x CC weight slab into LDS (async if available)
    {
        const _Float16* src = w1t + (size_t)n_base * CC;
        const int chunks = CC / 8;              // 24 b128 chunks per row
        for (int i = threadIdx.x; i < 128 * chunks; i += 256) {
            const int r = i / chunks, c = i - r * chunks;
            cp_b128(src + (size_t)r * CC + c * 8, &sB[r * KP1 + c * 8]);
        }
    }
    stage_wait();

    // A row gather with optional H/W transpose
    const int row = m_tile * 16 + (lane & 15);
    const int b   = row / LL;
    const int l   = row - b * LL;
    const int h   = l / WW, w = l - h * WW;
    const bool tv = isv[b] != 0;
    const int hh = tv ? w : h;
    const int ww = tv ? h : w;
    const _Float16* arow = xn + ((size_t)b * LL + (size_t)hh * WW + ww) * CC;

    const int ka = (lane >> 4) << 3;   // 0 or 8
    const int kb = (lane >> 4) << 3;   // 0 or 8
    const int nl = lane & 15;
    const _Float16* ap = arow + ka;

    v8f acc[8] = {v8f{}, v8f{}, v8f{}, v8f{}, v8f{}, v8f{}, v8f{}, v8f{}};
    v8h alo = *(const v8h*)(ap);
    v8h ahi = *(const v8h*)(ap + 16);
#pragma unroll
    for (int k0 = 0; k0 < 6; ++k0) {
        const v16h a = make_frag(alo, ahi);
        if (k0 < 5) {                            // prefetch next A fragment
            alo = *(const v8h*)(ap + (k0+1)*32);
            ahi = *(const v8h*)(ap + (k0+1)*32 + 16);
        }
        // gather ALL B fragments (one ds clause + one wait), then WMMA chain.
        v16h bf[8];
#pragma unroll
        for (int t = 0; t < 8; ++t)
            bf[t] = load_b_frag(sB, t*16 + nl, KP1, k0*32 + kb);
        SCHED_FENCE();
#pragma unroll
        for (int t = 0; t < 8; ++t)
            acc[t] = __builtin_amdgcn_wmma_f32_16x16x32_f16(
                false, a, false, bf[t], (short)0, acc[t], false, false);
    }

    // epilogue: +bias, store f16
    const int rr = (lane >> 4) << 3;
#pragma unroll
    for (int t = 0; t < 8; ++t) {
        const int n = n_base + t*16 + nl;
        const float bias = b1[n];
#pragma unroll
        for (int r = 0; r < 8; ++r) {
            const int m = m_tile * 16 + rr + r;
            t1[(size_t)m * DD + n] = (_Float16)(acc[t][r] + bias);
        }
    }
}

// ---------------- K4: dwconv+GELU fused into fc2 GEMM + residual ---------
// grid: (MT/8, 1); wave tile 16x192 (all of N) so GELU is computed ONCE.
__global__ void __launch_bounds__(256, 4)
fc2_kernel(const _Float16* __restrict__ t1,
           const _Float16* __restrict__ w2t,
           const float* __restrict__ b2,
           const float* __restrict__ dww,   // [D][3]
           const float* __restrict__ dwb,   // [D]
           const float* __restrict__ x,
           float* __restrict__ out) {
    const int lane   = threadIdx.x & 31;
    const int wave   = threadIdx.x >> 5;
    const int m_tile = blockIdx.x * 8 + wave;

    const int row = m_tile * 16 + (lane & 15);
    const int b   = row / LL;
    const int l   = row - b * LL;
    const _Float16* cur  = t1 + (size_t)row * DD;
    const _Float16* prev = (l > 0)      ? cur - DD : nullptr;
    const _Float16* next = (l < LL - 1) ? cur + DD : nullptr;

    const int ka = (lane >> 4) << 3;
    const int kb = (lane >> 4) << 3;
    const int nl = lane & 15;
    const v8h zero = {};

    // prefetched raw chunks for current k-step
    v8h pc[2], cc_[2], nc_[2];
#pragma unroll
    for (int chunk = 0; chunk < 2; ++chunk) {
        const int kk = ka + chunk * 16;
        cc_[chunk] = *(const v8h*)(cur + kk);
        pc[chunk]  = prev ? *(const v8h*)(prev + kk) : zero;
        nc_[chunk] = next ? *(const v8h*)(next + kk) : zero;
    }

    v8f acc[12];
#pragma unroll
    for (int t = 0; t < 12; ++t) acc[t] = v8f{};

#pragma unroll
    for (int k0 = 0; k0 < 12; ++k0) {
        // depthwise conv (k=3) + exact GELU -> A fragment (computed once)
        v16h a;
#pragma unroll
        for (int chunk = 0; chunk < 2; ++chunk) {
            const int kk = k0 * 32 + ka + chunk * 16;
#pragma unroll
            for (int j = 0; j < 8; ++j) {
                const int ch = kk + j;
                float v = (float)pc[chunk][j]  * dww[ch*3+0]
                        + (float)cc_[chunk][j] * dww[ch*3+1]
                        + (float)nc_[chunk][j] * dww[ch*3+2]
                        + dwb[ch];
                v = 0.5f * v * (1.0f + erff(v * 0.70710678118654752f));
                a[chunk*8 + j] = (_Float16)v;
            }
        }
        if (k0 < 11) {                           // prefetch next raw chunks
#pragma unroll
            for (int chunk = 0; chunk < 2; ++chunk) {
                const int kk = (k0+1) * 32 + ka + chunk * 16;
                cc_[chunk] = *(const v8h*)(cur + kk);
                pc[chunk]  = prev ? *(const v8h*)(prev + kk) : zero;
                nc_[chunk] = next ? *(const v8h*)(next + kk) : zero;
            }
        }
        // two batches of 6 B fragments: clause the loads, then WMMA chain
#pragma unroll
        for (int half = 0; half < 2; ++half) {
            v16h bf[6];
#pragma unroll
            for (int t = 0; t < 6; ++t)
                bf[t] = load_b_frag(w2t, (half*6 + t)*16 + nl, DD, k0*32 + kb);
            SCHED_FENCE();
#pragma unroll
            for (int t = 0; t < 6; ++t)
                acc[half*6 + t] = __builtin_amdgcn_wmma_f32_16x16x32_f16(
                    false, a, false, bf[t], (short)0, acc[half*6 + t],
                    false, false);
        }
    }

    // epilogue: +bias +residual, fp32 out (original flatten order)
    const int rr = (lane >> 4) << 3;
#pragma unroll
    for (int t = 0; t < 12; ++t) {
        const int n = t*16 + nl;
        const float bias = b2[n];
#pragma unroll
        for (int r = 0; r < 8; ++r) {
            const int m = m_tile * 16 + rr + r;
            const size_t o = (size_t)m * CC + n;
            out[o] = x[o] + acc[t][r] + bias;
        }
    }
}

// ---------------- host launcher ----------------
extern "C" void kernel_launch(void* const* d_in, const int* in_sizes, int n_in,
                              void* d_out, int out_size, void* d_ws, size_t ws_size,
                              hipStream_t stream) {
    const float* x      = (const float*)d_in[0];
    const float* gamma  = (const float*)d_in[1];
    const float* beta   = (const float*)d_in[2];
    const float* sel_w1 = (const float*)d_in[3];
    const float* sel_b1 = (const float*)d_in[4];
    const float* sel_w2 = (const float*)d_in[5];
    const float* sel_b2 = (const float*)d_in[6];
    const float* fc1_w  = (const float*)d_in[7];
    const float* fc1_b  = (const float*)d_in[8];
    const float* dw_w   = (const float*)d_in[9];
    const float* dw_b   = (const float*)d_in[10];
    const float* fc2_w  = (const float*)d_in[11];
    const float* fc2_b  = (const float*)d_in[12];
    float* out = (float*)d_out;

    char* ws = (char*)d_ws;
    _Float16* xn   = (_Float16*)(ws + OFF_XN);
    float*    gray = (float*)   (ws + OFF_GRAY);
    float*    gsum = (float*)   (ws + OFF_GSUM);
    int*      isv  = (int*)     (ws + OFF_ISV);
    _Float16* w1t  = (_Float16*)(ws + OFF_W1T);
    _Float16* w2t  = (_Float16*)(ws + OFF_W2T);
    _Float16* t1   = (_Float16*)(ws + OFF_T1);

    ln_kernel<<<MM/8, 256, 0, stream>>>(x, gamma, beta, xn, gray);

    (void)hipMemsetAsync(gsum, 0, BB * 2 * sizeof(float), stream);
    grad_kernel<<<dim3((LL + 255)/256, BB), 256, 0, stream>>>(gray, gsum);
    sel_kernel<<<1, 32, 0, stream>>>(gsum, sel_w1, sel_b1, sel_w2, sel_b2, isv);

    wt_kernel<<<(CC*DD + 255)/256, 256, 0, stream>>>(fc1_w, w1t, CC, DD);
    wt_kernel<<<(DD*CC + 255)/256, 256, 0, stream>>>(fc2_w, w2t, DD, CC);

    fc1_kernel<<<dim3(MT/8, DD/128), 256, 0, stream>>>(xn, w1t, fc1_b, isv, t1);
    fc2_kernel<<<MT/8, 256, 0, stream>>>(t1, w2t, fc2_b, dw_w, dw_b, x, out);
    (void)in_sizes; (void)n_in; (void)out_size; (void)ws_size;
}
